// Model_644245095219
// MI455X (gfx1250) — compile-verified, matrix operand order checked by reading, not run on hardware
//
#include <hip/hip_runtime.h>
#include <hip/hip_bf16.h>
#include <stdint.h>

// ---------------------------------------------------------------------------
// CDNA5 (gfx1250) WMMA types — wave32, 16x16x32 bf16 -> f32
// ---------------------------------------------------------------------------
typedef __attribute__((ext_vector_type(16))) __bf16 v16bf;
typedef __attribute__((ext_vector_type(8)))  float  v8f;

union FragBF16 {            // one 16x32 (A) or 32x16 (B) bf16 fragment: 16 bf16/lane
    v16bf bf;
    uint4 q[2];             // two 16-byte LDS/global reads
    uint32_t w[8];
    unsigned short u[16];
};

__device__ __forceinline__ unsigned short f2bf(float f) {
    uint32_t u = __builtin_bit_cast(uint32_t, f);
    uint32_t r = u + 0x7FFFu + ((u >> 16) & 1u);   // round-to-nearest-even
    return (unsigned short)(r >> 16);
}

// packed f32x2 -> bf16x2 (single v_cvt_pk_bf16_f32 when available)
__device__ __forceinline__ uint32_t pk2bf(float a, float b) {
#if defined(__AMDGCN__) && __has_builtin(__builtin_amdgcn_cvt_pk_bf16_f32)
    typedef __attribute__((ext_vector_type(2))) __bf16 v2bf;
    v2bf r = __builtin_amdgcn_cvt_pk_bf16_f32(a, b);
    return __builtin_bit_cast(uint32_t, r);
#else
    return (uint32_t)f2bf(a) | ((uint32_t)f2bf(b) << 16);
#endif
}

__device__ __forceinline__ v8f wmma_bf16(const FragBF16& a, const FragBF16& b, v8f c) {
    return __builtin_amdgcn_wmma_f32_16x16x32_bf16(
        /*neg_a=*/false, a.bf, /*neg_b=*/false, b.bf,
        /*c_mod=*/(short)0, c, /*reuse_a=*/false, /*reuse_b=*/false);
}

// ---------------------------------------------------------------------------
// GEMM: C[M,N] = A[M,K](f32,row-major) * B[N,K](f32,row-major)^T
// Tile 128x128, K-chunk 32, 256 threads = 8 waves (4 along M x 2 along N),
// each wave computes a 32x64 sub-tile = 8 WMMA accumulators.
// All global tile loads batched into registers before fp32->bf16 conversion,
// so the loads pipeline under a single s_wait_loadcnt.
// ---------------------------------------------------------------------------
#define GEMM_BM 128
#define GEMM_BN 128
#define GEMM_KC 32

__global__ __launch_bounds__(256)
void gemm_bf16_wmma(const float* __restrict__ A, const float* __restrict__ B,
                    float* __restrict__ C, int M, int N, int K) {
    __shared__ __align__(16) unsigned short sA[GEMM_BM * GEMM_KC]; // 8 KB
    __shared__ __align__(16) unsigned short sB[GEMM_BN * GEMM_KC]; // 8 KB

    const int t     = threadIdx.x;
    const int lane  = t & 31;
    const int wave  = t >> 5;        // 0..7
    const int waveM = wave & 3;      // 4 waves along M (32 rows each)
    const int waveN = wave >> 2;     // 2 waves along N (64 cols each)
    const int l16   = lane & 15;
    const int hs    = lane >> 4;     // lane half-select (K split per ISA layout)

    const int rowBase = blockIdx.x * GEMM_BM;
    const int colBase = blockIdx.y * GEMM_BN;

    v8f acc[2][4];
    const v8f vzero = {0.f, 0.f, 0.f, 0.f, 0.f, 0.f, 0.f, 0.f};
#pragma unroll
    for (int mi = 0; mi < 2; ++mi)
#pragma unroll
        for (int ni = 0; ni < 4; ++ni) acc[mi][ni] = vzero;

    const int c4 = (t & 7) * 4;      // 8 float4 columns cover KC=32
    const int r0 = t >> 3;           // 0..31 (rows stride 32)

    for (int kb = 0; kb < K; kb += GEMM_KC) {
        // ---- batched loads: 4x A rows + 4x B rows per thread ----
        float4 va[4], vb[4];
#pragma unroll
        for (int i = 0; i < 4; ++i) {
            va[i] = *(const float4*)(A + (size_t)(rowBase + r0 + 32 * i) * K + kb + c4);
            vb[i] = *(const float4*)(B + (size_t)(colBase + r0 + 32 * i) * K + kb + c4);
        }
        if (kb + GEMM_KC < K) {      // pull next A stripe toward L2
            __builtin_prefetch(A + (size_t)(rowBase + r0) * K + kb + GEMM_KC + c4, 0, 1);
        }
        // ---- convert + store to LDS (packed bf16x2 writes) ----
#pragma unroll
        for (int i = 0; i < 4; ++i) {
            uint32_t* da = (uint32_t*)&sA[(r0 + 32 * i) * GEMM_KC + c4];
            da[0] = pk2bf(va[i].x, va[i].y);
            da[1] = pk2bf(va[i].z, va[i].w);
            uint32_t* db = (uint32_t*)&sB[(r0 + 32 * i) * GEMM_KC + c4];
            db[0] = pk2bf(vb[i].x, vb[i].y);
            db[1] = pk2bf(vb[i].z, vb[i].w);
        }
        __syncthreads();

        // ---- load fragments per ISA layout: per lane two contiguous b128 ----
        FragBF16 fa[2], fb[4];
#pragma unroll
        for (int mi = 0; mi < 2; ++mi) {
            const unsigned short* p = &sA[(waveM * 32 + mi * 16 + l16) * GEMM_KC + hs * 8];
            fa[mi].q[0] = *(const uint4*)(p);
            fa[mi].q[1] = *(const uint4*)(p + 16);
        }
#pragma unroll
        for (int ni = 0; ni < 4; ++ni) {
            const unsigned short* p = &sB[(waveN * 64 + ni * 16 + l16) * GEMM_KC + hs * 8];
            fb[ni].q[0] = *(const uint4*)(p);
            fb[ni].q[1] = *(const uint4*)(p + 16);
        }
#pragma unroll
        for (int mi = 0; mi < 2; ++mi)
#pragma unroll
            for (int ni = 0; ni < 4; ++ni)
                acc[mi][ni] = wmma_bf16(fa[mi], fb[ni], acc[mi][ni]);
        __syncthreads();
    }

    // ---- store: C/D layout: VGPR r -> M = r + 8*hs ; N = l16 ----
#pragma unroll
    for (int mi = 0; mi < 2; ++mi)
#pragma unroll
        for (int ni = 0; ni < 4; ++ni) {
            const int col = colBase + waveN * 64 + ni * 16 + l16;
#pragma unroll
            for (int r = 0; r < 8; ++r) {
                const int row = rowBase + waveM * 32 + mi * 16 + hs * 8 + r;
                C[(size_t)row * N + col] = acc[mi][ni][r];
            }
        }
}

// ---------------------------------------------------------------------------
// Per-head RMSNorm + RoPE (in place). One wave per (seq, head) row of 128.
// Lane t owns dims {t, t+32, t+64, t+96} so rotate_half pairs stay lane-local.
// ---------------------------------------------------------------------------
__global__ __launch_bounds__(256)
void qk_norm_rope(float* __restrict__ X, const float* __restrict__ w,
                  const float* __restrict__ cosT, const float* __restrict__ sinT,
                  const int* __restrict__ pos, int nHeads, int S) {
    const int wid  = blockIdx.x * (blockDim.x >> 5) + (threadIdx.x >> 5);
    const int lane = threadIdx.x & 31;
    if (wid >= S * nHeads) return;
    const int s = wid / nHeads;
    const int h = wid - s * nHeads;

    float* row = X + (size_t)s * nHeads * 128 + (size_t)h * 128;
    float x0 = row[lane], x1 = row[lane + 32], x2 = row[lane + 64], x3 = row[lane + 96];

    float ss = x0 * x0 + x1 * x1 + x2 * x2 + x3 * x3;
#pragma unroll
    for (int m = 16; m >= 1; m >>= 1) ss += __shfl_xor(ss, m, 32);
    const float rs = rsqrtf(ss * (1.0f / 128.0f) + 1e-6f);

    x0 *= rs * w[lane];      x1 *= rs * w[lane + 32];
    x2 *= rs * w[lane + 64]; x3 *= rs * w[lane + 96];

    const int p = pos[s];
    const float* c  = cosT + (size_t)p * 128;
    const float* sn = sinT + (size_t)p * 128;
    row[lane]      = x0 * c[lane]      - x2 * sn[lane];
    row[lane + 32] = x1 * c[lane + 32] - x3 * sn[lane + 32];
    row[lane + 64] = x2 * c[lane + 64] + x0 * sn[lane + 64];
    row[lane + 96] = x3 * c[lane + 96] + x1 * sn[lane + 96];
}

// ---------------------------------------------------------------------------
// Causal flash attention, GQA (n_rep=4). Grid: (S/128, 32 heads), 256 threads.
// Each wave owns 16 query rows with private online-softmax state.
// QK^T and P*V both use v_wmma_f32_16x16x32_bf16. K-step = 32 keys.
// ---------------------------------------------------------------------------
#define AQ 128
#define AK 32

__global__ __launch_bounds__(256)
void flash_attn(const float* __restrict__ Q, const float* __restrict__ Kp,
                const float* __restrict__ Vp, float* __restrict__ O, int S) {
    __shared__ __align__(16) unsigned short sK[AK * 128];      // [key][dim]  8 KB
    __shared__ __align__(16) unsigned short sVt[128 * AK];     // [dim][key]  8 KB
    __shared__ __align__(16) unsigned short sP[8 * 16 * AK];   // per-wave P  8 KB

    const int t    = threadIdx.x;
    const int lane = t & 31;
    const int wave = t >> 5;       // 0..7 -> 16 query rows each
    const int l16  = lane & 15;
    const int hs   = lane >> 4;

    const int h   = blockIdx.y;    // query head
    const int kvh = h >> 2;        // GQA: 4 query heads share a KV head
    const int q0  = blockIdx.x * AQ;
    const int qRow = q0 + wave * 16 + l16;   // A-fragment row (M = l16)

    // ---- preload Q fragments (4 K-chunks of 32 dims), scaled by 1/sqrt(d) ----
    const float scale = 0.08838834764831845f;
    const float* qptr = Q + (size_t)qRow * 4096 + (size_t)h * 128;
    FragBF16 qf[4];
#pragma unroll
    for (int kc = 0; kc < 4; ++kc) {
        const float4 v0 = *(const float4*)(qptr + kc * 32 + hs * 8);
        const float4 v1 = *(const float4*)(qptr + kc * 32 + hs * 8 + 4);
        const float4 v2 = *(const float4*)(qptr + kc * 32 + 16 + hs * 8);
        const float4 v3 = *(const float4*)(qptr + kc * 32 + 16 + hs * 8 + 4);
        qf[kc].w[0] = pk2bf(v0.x * scale, v0.y * scale);
        qf[kc].w[1] = pk2bf(v0.z * scale, v0.w * scale);
        qf[kc].w[2] = pk2bf(v1.x * scale, v1.y * scale);
        qf[kc].w[3] = pk2bf(v1.z * scale, v1.w * scale);
        qf[kc].w[4] = pk2bf(v2.x * scale, v2.y * scale);
        qf[kc].w[5] = pk2bf(v2.z * scale, v2.w * scale);
        qf[kc].w[6] = pk2bf(v3.x * scale, v3.y * scale);
        qf[kc].w[7] = pk2bf(v3.z * scale, v3.w * scale);
    }

    // ---- flash state: rows r + 8*hs (C/D layout), replicated across 16 lanes ----
    const v8f vzero = {0.f, 0.f, 0.f, 0.f, 0.f, 0.f, 0.f, 0.f};
    v8f o[8];
    float rmax[8], rsum[8];
#pragma unroll
    for (int i = 0; i < 8; ++i) { o[i] = vzero; rmax[i] = -3.0e38f; rsum[i] = 0.f; }

    const int kEnd = q0 + AQ;                  // causal upper bound (exclusive)
    const int c4 = (t & 31) * 4;               // 32 float4 cover 128 dims
    const int rB = t >> 5;                     // 0..7, stride 8 -> 32 key rows

    for (int kb = 0; kb < kEnd; kb += AK) {
        // ---- batched K/V tile loads, then convert: K row-major, V transposed ----
        float4 kr[4], vr[4];
#pragma unroll
        for (int i = 0; i < 4; ++i) {
            const size_t base = (size_t)(kb + rB + 8 * i) * 1024 + (size_t)kvh * 128 + c4;
            kr[i] = *(const float4*)(Kp + base);
            vr[i] = *(const float4*)(Vp + base);
        }
#pragma unroll
        for (int i = 0; i < 4; ++i) {
            const int r = rB + 8 * i;
            uint32_t* dk = (uint32_t*)&sK[r * 128 + c4];
            dk[0] = pk2bf(kr[i].x, kr[i].y);
            dk[1] = pk2bf(kr[i].z, kr[i].w);
            sVt[(c4 + 0) * AK + r] = f2bf(vr[i].x);
            sVt[(c4 + 1) * AK + r] = f2bf(vr[i].y);
            sVt[(c4 + 2) * AK + r] = f2bf(vr[i].z);
            sVt[(c4 + 3) * AK + r] = f2bf(vr[i].w);
        }
        __syncthreads();

        // ---- S = Q K^T : 16 x 32 scores (2 tiles), K-dim 128 = 4 chunks ----
        v8f sc[2] = {vzero, vzero};
#pragma unroll
        for (int kc = 0; kc < 4; ++kc) {
            FragBF16 bk[2];
#pragma unroll
            for (int ni = 0; ni < 2; ++ni) {
                const unsigned short* p = &sK[(ni * 16 + l16) * 128 + kc * 32 + hs * 8];
                bk[ni].q[0] = *(const uint4*)(p);
                bk[ni].q[1] = *(const uint4*)(p + 16);
            }
            sc[0] = wmma_bf16(qf[kc], bk[0], sc[0]);
            sc[1] = wmma_bf16(qf[kc], bk[1], sc[1]);
        }

        // ---- causal mask + online softmax (row = 16-lane group reduce) ----
#pragma unroll
        for (int r = 0; r < 8; ++r) {
            const int qg = q0 + wave * 16 + hs * 8 + r;
            float s0 = (kb + l16      <= qg) ? sc[0][r] : -3.0e38f;
            float s1 = (kb + 16 + l16 <= qg) ? sc[1][r] : -3.0e38f;
            float mx = fmaxf(s0, s1);
#pragma unroll
            for (int m = 8; m >= 1; m >>= 1) mx = fmaxf(mx, __shfl_xor(mx, m, 32));
            const float newm = fmaxf(rmax[r], mx);
            const float corr = __expf(rmax[r] - newm);
            const float p0 = __expf(s0 - newm);
            const float p1 = __expf(s1 - newm);
            float psum = p0 + p1;
#pragma unroll
            for (int m = 8; m >= 1; m >>= 1) psum += __shfl_xor(psum, m, 32);
            rsum[r] = rsum[r] * corr + psum;
            rmax[r] = newm;
#pragma unroll
            for (int n = 0; n < 8; ++n) o[n][r] *= corr;
            // P to LDS (C-layout -> A-layout relayout; wave-private slab,
            // same-wave DS ops are in program order -> no barrier needed)
            sP[wave * 512 + (hs * 8 + r) * 32 + l16]      = f2bf(p0);
            sP[wave * 512 + (hs * 8 + r) * 32 + 16 + l16] = f2bf(p1);
        }

        // ---- O += P V : 16x128 output = 8 tiles along dims ----
        FragBF16 pf;
        {
            const unsigned short* pp = &sP[wave * 512 + l16 * 32 + hs * 8];
            pf.q[0] = *(const uint4*)(pp);
            pf.q[1] = *(const uint4*)(pp + 16);
        }
#pragma unroll
        for (int ni = 0; ni < 8; ++ni) {
            FragBF16 vf;
            const unsigned short* vp = &sVt[(ni * 16 + l16) * AK + hs * 8];
            vf.q[0] = *(const uint4*)(vp);
            vf.q[1] = *(const uint4*)(vp + 16);
            o[ni] = wmma_bf16(pf, vf, o[ni]);
        }
        __syncthreads();   // protect sK/sVt restage
    }

    // ---- epilogue: normalize and store [S, 32*128] f32 ----
#pragma unroll
    for (int ni = 0; ni < 8; ++ni) {
        const int d = ni * 16 + l16;
#pragma unroll
        for (int r = 0; r < 8; ++r) {
            const int qg = q0 + wave * 16 + hs * 8 + r;
            O[(size_t)qg * 4096 + (size_t)h * 128 + d] = o[ni][r] / rsum[r];
        }
    }
}

// ---------------------------------------------------------------------------
// Orchestration
// ---------------------------------------------------------------------------
extern "C" void kernel_launch(void* const* d_in, const int* in_sizes, int n_in,
                              void* d_out, int out_size, void* d_ws, size_t ws_size,
                              hipStream_t stream) {
    const float* hidden = (const float*)d_in[0];
    const int*   pos    = (const int*)d_in[1];
    const float* Wq     = (const float*)d_in[2];
    const float* Wk     = (const float*)d_in[3];
    const float* Wv     = (const float*)d_in[4];
    const float* Wo     = (const float*)d_in[5];
    const float* qw     = (const float*)d_in[6];
    const float* kw     = (const float*)d_in[7];
    const float* cosT   = (const float*)d_in[8];
    const float* sinT   = (const float*)d_in[9];
    float* out = (float*)d_out;

    const int S = 2048, H = 4096, QD = 4096, KD = 1024;

    float* Qb = (float*)d_ws;                // [S, 4096]
    float* Kb = Qb + (size_t)S * QD;         // [S, 1024]
    float* Vb = Kb + (size_t)S * KD;         // [S, 1024]
    float* Ab = Vb + (size_t)S * KD;         // [S, 4096]

    const dim3 blk(256);
    // Projections (WMMA GEMMs)
    gemm_bf16_wmma<<<dim3(S / GEMM_BM, QD / GEMM_BN), blk, 0, stream>>>(hidden, Wq, Qb, S, QD, H);
    gemm_bf16_wmma<<<dim3(S / GEMM_BM, KD / GEMM_BN), blk, 0, stream>>>(hidden, Wk, Kb, S, KD, H);
    gemm_bf16_wmma<<<dim3(S / GEMM_BM, KD / GEMM_BN), blk, 0, stream>>>(hidden, Wv, Vb, S, KD, H);
    // QK RMSNorm + RoPE (one wave per row)
    qk_norm_rope<<<dim3((S * 32) / 8), blk, 0, stream>>>(Qb, qw, cosT, sinT, pos, 32, S);
    qk_norm_rope<<<dim3((S * 8) / 8),  blk, 0, stream>>>(Kb, kw, cosT, sinT, pos, 8,  S);
    // Causal GQA flash attention (WMMA)
    flash_attn<<<dim3(S / AQ, 32), blk, 0, stream>>>(Qb, Kb, Vb, Ab, S);
    // Output projection (WMMA GEMM)
    gemm_bf16_wmma<<<dim3(S / GEMM_BM, QD / GEMM_BN), blk, 0, stream>>>(Ab, Wo, out, S, QD, H);
}
